// gatedFusion_77970836291822
// MI455X (gfx1250) — matrix operand order repaired and checked by reading, not compiled, not run on hardware
//
#include <hip/hip_runtime.h>

#define NN 6144
#define TT 3072
#define DD 128
#define HH 4

typedef __attribute__((ext_vector_type(16))) __bf16 v16bf;
typedef __attribute__((ext_vector_type(8)))  float  v8f;

union BF16Frag {
  v16bf v;
  unsigned short u[16];
  float4 f4[2];
};

__device__ __forceinline__ unsigned short f2bf(float f) {
  unsigned int u = __float_as_uint(f);
  u += 0x7FFFu + ((u >> 16) & 1u);          // round-to-nearest-even
  return (unsigned short)(u >> 16);
}
__device__ __forceinline__ float bf2f(unsigned short b) {
  return __uint_as_float(((unsigned int)b) << 16);
}

// ---------------------------------------------------------------------------
// CDNA5 async global->LDS copies (ASYNCcnt) + wait.  LDS operand is the low
// 32 bits of the generic pointer (LDS aperture truncates to addr[31:0]).
// ---------------------------------------------------------------------------
__device__ __forceinline__ void async_load_b128(void* lds, const void* gaddr) {
  unsigned l = (unsigned)(size_t)lds;
  unsigned long long g = (unsigned long long)(size_t)gaddr;
  asm volatile("global_load_async_to_lds_b128 %0, %1, off" :: "v"(l), "v"(g) : "memory");
}
__device__ __forceinline__ void async_load_b32(void* lds, const void* gaddr) {
  unsigned l = (unsigned)(size_t)lds;
  unsigned long long g = (unsigned long long)(size_t)gaddr;
  asm volatile("global_load_async_to_lds_b32 %0, %1, off" :: "v"(l), "v"(g) : "memory");
}
__device__ __forceinline__ void wait_async() {
  asm volatile("s_wait_asynccnt 0x0" ::: "memory");
}

// ---------------------------------------------------------------------------
// f32 -> bf16 flat convert
// ---------------------------------------------------------------------------
__global__ void cvt_f32_bf16(const float* __restrict__ in,
                             unsigned short* __restrict__ out, int total) {
  int i = blockIdx.x * 256 + threadIdx.x;
  if (i < total) out[i] = f2bf(in[i]);
}

// ---------------------------------------------------------------------------
// [R,C] f32 -> [C,R] bf16 tiled transpose (batched). Block (32,8).
// ---------------------------------------------------------------------------
__global__ __launch_bounds__(256) void transpose_cvt(
    const float* __restrict__ in, unsigned short* __restrict__ out,
    int R, int C, long long iStride, long long oStride) {
  __shared__ float tile[32][33];
  const int bx = blockIdx.x * 32;   // C offset
  const int by = blockIdx.y * 32;   // R offset
  in  += (size_t)blockIdx.z * iStride;
  out += (size_t)blockIdx.z * oStride;
  const int tx = threadIdx.x, ty = threadIdx.y;
#pragma unroll
  for (int i = 0; i < 32; i += 8)
    tile[ty + i][tx] = in[(size_t)(by + ty + i) * C + bx + tx];
  __syncthreads();
#pragma unroll
  for (int i = 0; i < 32; i += 8)
    out[(size_t)(bx + ty + i) * R + by + tx] = f2bf(tile[tx][ty + i]);
}

// ---------------------------------------------------------------------------
// Generic bf16 WMMA GEMM, N fixed at 128.
//   A : [M,K] bf16 row-major
//   BT: [128,K] bf16 (i.e. B transposed, contiguous K-runs per output column)
// 4 waves/block, each wave owns a 16x128 tile. B tile double-buffered in LDS
// via async global->LDS copies; one barrier per K-step.
// MODE 0: store bf16 transposed [128,M]
// MODE 1: bias+relu, store bf16 transposed [128,M]
// MODE 2: bias+relu, store f32 row-major [M,128]
// MODE 3: plain store f32 row-major [M,128]
// ---------------------------------------------------------------------------
template <int MODE>
__global__ __launch_bounds__(128) void gemm_bf16_n128(
    const unsigned short* __restrict__ A,
    const unsigned short* __restrict__ BT,
    const float* __restrict__ bias,
    void* __restrict__ outp,
    int M, int K,
    long long aStride, long long bStride, long long oStride) {
  __shared__ unsigned short sB[2][128 * 48];  // 32 K-halves per row, padded to 48
  const int tid  = threadIdx.x;
  const int wave = tid >> 5;
  const int lane = tid & 31;
  const int hi   = (lane >> 4) & 1;
  const int r    = lane & 15;
  const int i0   = blockIdx.x * 64 + wave * 16;
  const int bat  = blockIdx.y;
  A  += (size_t)bat * aStride;
  BT += (size_t)bat * bStride;

  auto stageB = [&](int kc, int buf) {
    const unsigned short* gp = BT + (size_t)tid * K + kc;
#pragma unroll
    for (int q = 0; q < 4; ++q)
      async_load_b128(&sB[buf][tid * 48 + q * 8], gp + q * 8);
  };

  v8f acc[8] = {};
  stageB(0, 0);
  wait_async();
  __syncthreads();

  int buf = 0;
  for (int k0 = 0; k0 < K; k0 += 32) {
    if (k0 + 32 < K) stageB(k0 + 32, buf ^ 1);   // async into alternate buffer

    // A fragment: lane holds row (lane&15), K-halves {k0+8*hi..+7, k0+16+8*hi..+7}
    BF16Frag a;
    const unsigned short* arow = A + (size_t)(i0 + r) * K + k0 + hi * 8;
    a.f4[0] = *(const float4*)(arow);
    a.f4[1] = *(const float4*)(arow + 16);

#pragma unroll
    for (int t = 0; t < 8; ++t) {
      BF16Frag bfr;
      const unsigned short* bp = &sB[buf][(t * 16 + r) * 48 + hi * 8];
      bfr.f4[0] = *(const float4*)bp;
      bfr.f4[1] = *(const float4*)(bp + 16);
      acc[t] = __builtin_amdgcn_wmma_f32_16x16x32_bf16(
          false, a.v, false, bfr.v, (short)0, acc[t], false, false);
    }

    wait_async();
    __syncthreads();
    buf ^= 1;
  }

  if (MODE <= 1) {
    unsigned short* outT = (unsigned short*)outp + (size_t)bat * oStride;
#pragma unroll
    for (int t = 0; t < 8; ++t) {
      int c = t * 16 + r;
      float bv = (MODE == 1) ? bias[c] : 0.0f;
      alignas(16) unsigned short tmp[8];
#pragma unroll
      for (int v = 0; v < 8; ++v) {
        float val = acc[t][v] + bv;
        if (MODE == 1) val = fmaxf(val, 0.0f);
        tmp[v] = f2bf(val);
      }
      *(float4*)(outT + (size_t)c * M + i0 + hi * 8) = *(float4*)tmp;
    }
  } else {
    float* outR = (float*)outp + (size_t)bat * oStride;
#pragma unroll
    for (int t = 0; t < 8; ++t) {
      int c = t * 16 + r;
      float bv = (MODE == 2) ? bias[c] : 0.0f;
#pragma unroll
      for (int v = 0; v < 8; ++v) {
        float val = acc[t][v] + bv;
        if (MODE == 2) val = fmaxf(val, 0.0f);
        outR[(size_t)(i0 + v + hi * 8) * 128 + c] = val;
      }
    }
  }
}

// ---------------------------------------------------------------------------
// f_s[h,n] = sum_d WhbT[h][d][n]*a_src[h][d] ; likewise f_d
// ---------------------------------------------------------------------------
__global__ __launch_bounds__(256) void fsd_kernel(
    const unsigned short* __restrict__ WhbT,
    const float* __restrict__ a_src, const float* __restrict__ a_dst,
    float* __restrict__ f_s, float* __restrict__ f_d, int n) {
  int h = blockIdx.y;
  int i = blockIdx.x * 256 + threadIdx.x;
  const unsigned short* base = WhbT + (size_t)h * 128 * n + i;
  float ss = 0.0f, sd = 0.0f;
#pragma unroll 4
  for (int d = 0; d < 128; ++d) {
    float w = bf2f(base[(size_t)d * n]);
    ss += w * a_src[h * 128 + d];
    sd += w * a_dst[h * 128 + d];
  }
  f_s[(size_t)h * n + i] = ss;
  f_d[(size_t)h * n + i] = sd;
}

// ---------------------------------------------------------------------------
// Fused masked-softmax attention (flash style) + ELU epilogue.
// Block = 128 threads = 4 waves; wave w handles head w for one 16-row tile.
// adj/f_d tiles double-buffered in LDS via async copies, shared by all heads;
// next-next adj tile prefetched. P@V via bf16 WMMA, V from transposed WhbT.
// ---------------------------------------------------------------------------
__global__ __launch_bounds__(128) void attn_kernel(
    const unsigned short* __restrict__ WhbT,
    const float* __restrict__ f_s, const float* __restrict__ f_d,
    const int* __restrict__ adj,
    unsigned short* __restrict__ hcat, int n) {
  __shared__ int   s_adj[2][16 * 32];
  __shared__ float s_fd[2][HH * 32];
  const int tid  = threadIdx.x;
  const int h    = tid >> 5;
  const int lane = tid & 31;
  const int hi   = (lane >> 4) & 1;
  const int r    = lane & 15;
  const int i0   = blockIdx.x * 16;
  const unsigned short* V = WhbT + (size_t)h * 128 * n;
  const float fs_r = f_s[(size_t)h * n + i0 + r];
  float m = -3.0e38f, l = 0.0f;
  v8f acc[8] = {};

  const int rr = tid >> 3, cc = (tid & 7) * 4;   // coop adj-tile coords
  auto issue_tile = [&](int jb, int buf) {
    async_load_b128(&s_adj[buf][rr * 32 + cc],
                    &adj[(size_t)(i0 + rr) * n + jb + cc]);
    async_load_b32(&s_fd[buf][tid],
                   &f_d[(size_t)(tid >> 5) * n + jb + (tid & 31)]);
  };

  issue_tile(0, 0);
  wait_async();
  __syncthreads();

  int buf = 0;
  for (int j0 = 0; j0 < n; j0 += 32) {
    if (j0 + 32 < n) issue_tile(j0 + 32, buf ^ 1);
    if (j0 + 64 < n)
      __builtin_prefetch(&adj[(size_t)(i0 + rr) * n + j0 + 64 + cc], 0, 0);

    const float* fd0 = &s_fd[buf][h * 32 + hi * 8];
    const int*   ad0 = &s_adj[buf][r * 32 + hi * 8];
    float sv[16];
    float tmax = -3.0e38f;
#pragma unroll
    for (int e = 0; e < 16; ++e) {
      int off = (e < 8) ? e : (e + 8);           // K-halves: [0..7] and [16..23]
      float sc = fs_r + fd0[off];
      sc = (sc > 0.0f) ? sc : 0.2f * sc;         // leaky relu, alpha=0.2
      if (ad0[off] == 0) sc = -9.0e15f;          // mask
      sv[e] = sc;
      tmax = fmaxf(tmax, sc);
    }
    tmax = fmaxf(tmax, __shfl_xor(tmax, 16));    // row split across lane pair
    float newm  = fmaxf(m, tmax);
    float scale = __expf(m - newm);
    m = newm;
    float rs = 0.0f;
    BF16Frag a;
#pragma unroll
    for (int e = 0; e < 16; ++e) {
      float p = __expf(sv[e] - m);
      rs += p;
      a.u[e] = f2bf(p);
    }
    rs += __shfl_xor(rs, 16);
    l = l * scale + rs;

    float scv[8];
#pragma unroll
    for (int v = 0; v < 8; ++v) scv[v] = __shfl(scale, v + hi * 8);
#pragma unroll
    for (int t = 0; t < 8; ++t)
#pragma unroll
      for (int v = 0; v < 8; ++v) acc[t][v] *= scv[v];

#pragma unroll
    for (int t = 0; t < 8; ++t) {
      BF16Frag bfr;
      const unsigned short* bp = V + (size_t)(t * 16 + r) * n + j0 + hi * 8;
      bfr.f4[0] = *(const float4*)bp;
      bfr.f4[1] = *(const float4*)(bp + 16);
      acc[t] = __builtin_amdgcn_wmma_f32_16x16x32_bf16(
          false, a.v, false, bfr.v, (short)0, acc[t], false, false);
    }

    wait_async();
    __syncthreads();
    buf ^= 1;
  }

  float linv[8];
#pragma unroll
  for (int v = 0; v < 8; ++v) linv[v] = 1.0f / __shfl(l, v + hi * 8);
#pragma unroll
  for (int t = 0; t < 8; ++t) {
    int c = t * 16 + r;
#pragma unroll
    for (int v = 0; v < 8; ++v) {
      float val = acc[t][v] * linv[v];
      val = (val > 0.0f) ? val : (__expf(val) - 1.0f);   // ELU
      hcat[(size_t)(i0 + v + hi * 8) * (HH * DD) + h * DD + c] = f2bf(val);
    }
  }
}

// ---------------------------------------------------------------------------
// Gated fusion: z = sigmoid(c + t); f = z*c + (1-z)*t  (bf16 out for final GEMM)
// ---------------------------------------------------------------------------
__global__ void fuse_kernel(const float* __restrict__ cpt,
                            const float* __restrict__ ctx,
                            unsigned short* __restrict__ fus, int total) {
  int i = blockIdx.x * 256 + threadIdx.x;
  if (i < total) {
    float c = cpt[i], t = ctx[i];
    float z = 1.0f / (1.0f + __expf(-(c + t)));
    fus[i] = f2bf(z * c + (1.0f - z) * t);
  }
}

// ---------------------------------------------------------------------------
extern "C" void kernel_launch(void* const* d_in, const int* in_sizes, int n_in,
                              void* d_out, int out_size, void* d_ws, size_t ws_size,
                              hipStream_t stream) {
  (void)in_sizes; (void)n_in; (void)out_size; (void)ws_size;
  const float* x      = (const float*)d_in[0];
  const int*   adj    = (const int*)  d_in[1];
  const float* t_x    = (const float*)d_in[2];
  const int*   t_adj  = (const int*)  d_in[3];
  const float* tfidf  = (const float*)d_in[4];
  const float* Wg     = (const float*)d_in[5];
  const float* ag_src = (const float*)d_in[6];
  const float* ag_dst = (const float*)d_in[7];
  const float* fcg_W  = (const float*)d_in[8];
  const float* fcg_b  = (const float*)d_in[9];
  const float* Wt     = (const float*)d_in[10];
  const float* at_src = (const float*)d_in[11];
  const float* at_dst = (const float*)d_in[12];
  const float* fct_W  = (const float*)d_in[13];
  const float* fct_b  = (const float*)d_in[14];
  const float* fus_W  = (const float*)d_in[15];
  const float* fus_b  = (const float*)d_in[16];
  float* out = (float*)d_out;

  char* ws = (char*)d_ws;
  size_t off = 0;
  auto alloc = [&](size_t bytes) {
    size_t o = off; off += (bytes + 255) & ~(size_t)255; return o;
  };
  unsigned short* tfidfT = (unsigned short*)(ws + alloc((size_t)NN * TT * 2));
  unsigned short* xb     = (unsigned short*)(ws + alloc((size_t)NN * DD * 2));
  unsigned short* txb    = (unsigned short*)(ws + alloc((size_t)TT * DD * 2));
  unsigned short* WgT    = (unsigned short*)(ws + alloc((size_t)HH * DD * DD * 2));
  unsigned short* WtT    = (unsigned short*)(ws + alloc((size_t)HH * DD * DD * 2));
  unsigned short* fcgWT  = (unsigned short*)(ws + alloc((size_t)DD * HH * DD * 2));
  unsigned short* fctWT  = (unsigned short*)(ws + alloc((size_t)DD * HH * DD * 2));
  unsigned short* fusWT  = (unsigned short*)(ws + alloc((size_t)DD * DD * 2));
  unsigned short* WhbTg  = (unsigned short*)(ws + alloc((size_t)HH * DD * NN * 2));
  unsigned short* WhbTt  = (unsigned short*)(ws + alloc((size_t)HH * DD * TT * 2));
  float* fsg             = (float*)(ws + alloc((size_t)HH * NN * 4));
  float* fdg             = (float*)(ws + alloc((size_t)HH * NN * 4));
  float* fst             = (float*)(ws + alloc((size_t)HH * TT * 4));
  float* fdt             = (float*)(ws + alloc((size_t)HH * TT * 4));
  unsigned short* hcatg  = (unsigned short*)(ws + alloc((size_t)NN * HH * DD * 2));
  unsigned short* hcatt  = (unsigned short*)(ws + alloc((size_t)TT * HH * DD * 2));
  float* concept         = (float*)(ws + alloc((size_t)NN * DD * 4));
  unsigned short* gtT    = (unsigned short*)(ws + alloc((size_t)DD * TT * 2));
  float* ctext           = (float*)(ws + alloc((size_t)NN * DD * 4));
  unsigned short* fusion = (unsigned short*)(ws + alloc((size_t)NN * DD * 2));

  dim3 tb(32, 8);
  // ---- weight / input conversion & transposition (B operands pre-transposed)
  cvt_f32_bf16<<<(NN * DD + 255) / 256, 256, 0, stream>>>(x, xb, NN * DD);
  cvt_f32_bf16<<<(TT * DD + 255) / 256, 256, 0, stream>>>(t_x, txb, TT * DD);
  transpose_cvt<<<dim3(DD / 32, DD / 32, HH), tb, 0, stream>>>(Wg, WgT, DD, DD, DD * DD, DD * DD);
  transpose_cvt<<<dim3(DD / 32, DD / 32, HH), tb, 0, stream>>>(Wt, WtT, DD, DD, DD * DD, DD * DD);
  transpose_cvt<<<dim3(DD / 32, (HH * DD) / 32, 1), tb, 0, stream>>>(fcg_W, fcgWT, HH * DD, DD, 0, 0);
  transpose_cvt<<<dim3(DD / 32, (HH * DD) / 32, 1), tb, 0, stream>>>(fct_W, fctWT, HH * DD, DD, 0, 0);
  transpose_cvt<<<dim3(DD / 32, DD / 32, 1), tb, 0, stream>>>(fus_W, fusWT, DD, DD, 0, 0);
  transpose_cvt<<<dim3(NN / 32, TT / 32, 1), tb, 0, stream>>>(tfidf, tfidfT, TT, NN, 0, 0);

  // ---- Wh = x @ W[h]  (batched over heads, output transposed for later use)
  gemm_bf16_n128<0><<<dim3(NN / 64, HH), 128, 0, stream>>>(
      xb, WgT, nullptr, WhbTg, NN, DD, 0, (long long)DD * DD, (long long)DD * NN);
  gemm_bf16_n128<0><<<dim3(TT / 64, HH), 128, 0, stream>>>(
      txb, WtT, nullptr, WhbTt, TT, DD, 0, (long long)DD * DD, (long long)DD * TT);

  // ---- attention logits components
  fsd_kernel<<<dim3(NN / 256, HH), 256, 0, stream>>>(WhbTg, ag_src, ag_dst, fsg, fdg, NN);
  fsd_kernel<<<dim3(TT / 256, HH), 256, 0, stream>>>(WhbTt, at_src, at_dst, fst, fdt, TT);

  // ---- fused masked softmax attention + ELU -> concatenated head features
  attn_kernel<<<NN / 16, 128, 0, stream>>>(WhbTg, fsg, fdg, adj, hcatg, NN);
  attn_kernel<<<TT / 16, 128, 0, stream>>>(WhbTt, fst, fdt, t_adj, hcatt, TT);

  // ---- per-graph fc (Linear + ReLU)
  gemm_bf16_n128<2><<<dim3(NN / 64, 1), 128, 0, stream>>>(
      hcatg, fcgWT, fcg_b, concept, NN, HH * DD, 0, 0, 0);
  gemm_bf16_n128<1><<<dim3(TT / 64, 1), 128, 0, stream>>>(
      hcatt, fctWT, fct_b, gtT, TT, HH * DD, 0, 0, 0);   // transposed bf16 out

  // ---- c_text = tfidf^T @ gat_text
  gemm_bf16_n128<3><<<dim3(NN / 64, 1), 128, 0, stream>>>(
      tfidfT, gtT, nullptr, ctext, NN, TT, 0, 0, 0);

  // ---- sigmoid-gated fusion
  fuse_kernel<<<(NN * DD + 255) / 256, 256, 0, stream>>>(concept, ctext, fusion, NN * DD);

  // ---- final Linear + ReLU
  gemm_bf16_n128<2><<<dim3(NN / 64, 1), 128, 0, stream>>>(
      fusion, fusWT, fus_b, out, NN, DD, 0, 0, 0);
}
